// Discriminator_18743237280213
// MI455X (gfx1250) — matrix operand order repaired and checked by reading, not compiled, not run on hardware
//
#include <hip/hip_runtime.h>
#include <hip/hip_bf16.h>
#include <math.h>

typedef __attribute__((ext_vector_type(16))) _Float16 v16h;
typedef __attribute__((ext_vector_type(8)))  float    v8f;

#define LEAK 0.01f

__device__ __forceinline__ float lrelu(float x) { return x >= 0.f ? x : LEAK * x; }
// fast sigmoid/tanh on v_exp_f32 + v_rcp_f32 (no IEEE division sequences)
__device__ __forceinline__ float fsigm(float x)
{
    return __builtin_amdgcn_rcpf(1.f + __expf(-x));
}
__device__ __forceinline__ float ftanh(float x)
{
    float e = __expf(-2.f * x);
    return (1.f - e) * __builtin_amdgcn_rcpf(1.f + e);
}

// ---------------------------------------------------------------------------
// Kernel 1: fused conv1d(K=5)+lrelu+maxpool3 chain, one sample per block.
// Input  din : [4096][512][3]  (b*1536 + l*3 + p)
// Output x3  : [4096][18][10]  (b*180 + s*10 + c)  -- LSTM-ready layout
// Loops run co-outer / position-inner: no divisions, weights live in regs.
// ---------------------------------------------------------------------------
__global__ void conv_chain_kernel(const float* __restrict__ din,
                                  const float* __restrict__ W1, const float* __restrict__ b1,
                                  const float* __restrict__ W2, const float* __restrict__ b2,
                                  const float* __restrict__ W3, const float* __restrict__ b3,
                                  float* __restrict__ x3)
{
    __shared__ float s_in[3 * 512];   // input  [p][l]
    __shared__ float s_x[10 * 508];   // conv scratch (largest stage)
    __shared__ float s_y[10 * 170];   // pool scratch
    __shared__ float s_w[675];        // W1(150) b1(10) W2(250) b2(5) W3(250) b3(10)

    const int b   = blockIdx.x;
    const int tid = threadIdx.x;      // blockDim.x == 128

    // hint the prefetcher (global_prefetch_b8)
    __builtin_prefetch(din + b * 1536 + tid * 12, 0, 1);

    // stage weights
    for (int i = tid; i < 150; i += 128) s_w[i]       = W1[i];
    for (int i = tid; i < 10;  i += 128) s_w[150 + i] = b1[i];
    for (int i = tid; i < 250; i += 128) s_w[160 + i] = W2[i];
    for (int i = tid; i < 5;   i += 128) s_w[410 + i] = b2[i];
    for (int i = tid; i < 250; i += 128) s_w[415 + i] = W3[i];
    for (int i = tid; i < 10;  i += 128) s_w[665 + i] = b3[i];

    // stage input: [l][p] -> [p][l] (coalesced global read)
    for (int i = tid; i < 1536; i += 128) {
        int l = i / 3, p = i - l * 3;
        s_in[p * 512 + l] = din[(size_t)b * 1536 + i];
    }
    __syncthreads();

    // conv1: 3->10, Lin=512, Lout=508
    for (int co = 0; co < 10; ++co) {
        float w[15];
        #pragma unroll
        for (int t = 0; t < 15; ++t) w[t] = s_w[co * 15 + t];
        const float bb = s_w[150 + co];
        for (int o = tid; o < 508; o += 128) {
            float acc = bb;
            #pragma unroll
            for (int ci = 0; ci < 3; ++ci)
                #pragma unroll
                for (int k = 0; k < 5; ++k)
                    acc += w[ci * 5 + k] * s_in[ci * 512 + o + k];
            s_x[co * 508 + o] = lrelu(acc);
        }
    }
    __syncthreads();
    // pool1: 508 -> 170 (pad=1, -inf)
    for (int co = 0; co < 10; ++co)
        for (int o = tid; o < 170; o += 128) {
            float m = -INFINITY;
            #pragma unroll
            for (int d = 0; d < 3; ++d) {
                int l = 3 * o - 1 + d;
                if (l >= 0 && l < 508) m = fmaxf(m, s_x[co * 508 + l]);
            }
            s_y[co * 170 + o] = m;
        }
    __syncthreads();
    // conv2: 10->5, Lin=170, Lout=166
    for (int co = 0; co < 5; ++co) {
        float w[50];
        #pragma unroll
        for (int t = 0; t < 50; ++t) w[t] = s_w[160 + co * 50 + t];
        const float bb = s_w[410 + co];
        for (int o = tid; o < 166; o += 128) {
            float acc = bb;
            #pragma unroll
            for (int ci = 0; ci < 10; ++ci)
                #pragma unroll
                for (int k = 0; k < 5; ++k)
                    acc += w[ci * 5 + k] * s_y[ci * 170 + o + k];
            s_x[co * 166 + o] = lrelu(acc);
        }
    }
    __syncthreads();
    // pool2: 166 -> 56
    for (int co = 0; co < 5; ++co)
        for (int o = tid; o < 56; o += 128) {
            float m = -INFINITY;
            #pragma unroll
            for (int d = 0; d < 3; ++d) {
                int l = 3 * o - 1 + d;
                if (l >= 0 && l < 166) m = fmaxf(m, s_x[co * 166 + l]);
            }
            s_y[co * 56 + o] = m;
        }
    __syncthreads();
    // conv3: 5->10, Lin=56, Lout=52
    for (int co = 0; co < 10; ++co) {
        float w[25];
        #pragma unroll
        for (int t = 0; t < 25; ++t) w[t] = s_w[415 + co * 25 + t];
        const float bb = s_w[665 + co];
        for (int o = tid; o < 52; o += 128) {
            float acc = bb;
            #pragma unroll
            for (int ci = 0; ci < 5; ++ci)
                #pragma unroll
                for (int k = 0; k < 5; ++k)
                    acc += w[ci * 5 + k] * s_y[ci * 56 + o + k];
            s_x[co * 52 + o] = lrelu(acc);
        }
    }
    __syncthreads();
    // pool3: 52 -> 18, write [s][c] to global
    for (int i = tid; i < 180; i += 128) {
        int co = i / 18, o = i - co * 18;
        float m = -INFINITY;
        #pragma unroll
        for (int d = 0; d < 3; ++d) {
            int l = 3 * o - 1 + d;
            if (l >= 0 && l < 52) m = fmaxf(m, s_x[co * 52 + l]);
        }
        x3[(size_t)b * 180 + o * 10 + co] = m;
    }
}

// ---------------------------------------------------------------------------
// Kernel 2: 3-layer LSTM (H=10), one wave32 per 16-sample tile.
// Gate GEMM G[16,40] = [X_t|H](16x20, K-padded to 32) x W(20x40) via
// v_wmma_f32_16x16x32_f16 (3 N-tiles of 16), bias pre-loaded in the f32
// accumulator. D tiles kept in fragment order so each stores as 2x
// ds_store_b128; all elementwise addressing is hoisted out of the step loop.
// xin: [B][T][10], out: [B][10] = h_last(L2)+h_last(L1). grid = B/16, 32 thr.
// ---------------------------------------------------------------------------
__global__ void lstm3_kernel(const float* __restrict__ xin,
                             const float* __restrict__ Wih,   // [3][40][10]
                             const float* __restrict__ Whh,   // [3][40][10]
                             const float* __restrict__ bias,  // [3][40]
                             float* __restrict__ out,         // [B][10]
                             int T)
{
    extern __shared__ __align__(16) float smem[];
    const int TT10 = T * 10;
    float* xbuf  = smem;                 // [16][T][10] layer input (rewritten in place)
    float* Hs    = xbuf + 16 * TT10;     // [16][10]
    float* Cs    = Hs + 160;             // [16][10]
    float* G     = Cs + 160;             // fragment order: [32 lanes][24]
    float* hprev = G + 768;              // [16][10] layer-1 final h

    const int lane = threadIdx.x;        // full wave active -> EXEC all-ones at WMMA
    const int tile = blockIdx.x;
    const int m    = lane & 15;
    const int hi   = lane >> 4;

    // stage the 16-sample input tile (no divisions)
    {
        const float* src = xin + (size_t)tile * 16 * TT10;
        for (int bl = 0; bl < 16; ++bl)
            for (int i = lane; i < TT10; i += 32)
                xbuf[bl * TT10 + i] = src[bl * TT10 + i];
    }
    __syncthreads();

    const float* xrow = xbuf + m * TT10;
    const float* hrow = Hs + m * 10;
    float*       gfr  = G + lane * 24;

    // ---- step-invariant elementwise addressing (5 items per lane) ----------
    int gI[5], gF[5], gG[5], gO[5];      // G indices for gates i,f,g,o
    int xb[5];                           // xbuf base (add s*10 per step)
    int ob[5];                           // output index
    #pragma unroll
    for (int q = 0; q < 5; ++q) {
        int p  = lane + q * 32;
        int bl = p / 10, u = p - bl * 10;
        int lh = (bl >> 3) << 4;         // 0 or 16
        int r  = bl & 7;
        int ni = u, nf = 10 + u, ng = 20 + u, no = 30 + u;
        gI[q] = ((ni & 15) + lh) * 24 + ((ni >> 4) << 3) + r;
        gF[q] = ((nf & 15) + lh) * 24 + ((nf >> 4) << 3) + r;
        gG[q] = ((ng & 15) + lh) * 24 + ((ng >> 4) << 3) + r;
        gO[q] = ((no & 15) + lh) * 24 + ((no >> 4) << 3) + r;
        xb[q] = bl * TT10 + u;
        ob[q] = (tile * 16 + bl) * 10 + u;
    }

    for (int l = 0; l < 3; ++l) {
        const float* wih = Wih  + l * 400;
        const float* whh = Whh  + l * 400;
        const float* bb  = bias + l * 40;

        // B fragments: B[k][n] = k<10 ? Wih[n][k] : (k<20 ? Whh[n][k-10] : 0)
        // 16-bit B layout (32x16): lane<16 holds K=0..7 / 16..23 (N=lane&15), lane>=16 holds K=8..15 / 24..31
        v16h  bf[3];
        float bias_n[3];
        const int khalf = hi * 8;
        #pragma unroll
        for (int nt = 0; nt < 3; ++nt) {
            const int n = m + nt * 16;
            #pragma unroll
            for (int j = 0; j < 16; ++j) {
                const int k = khalf + j + (j & 8);
                float v = 0.f;
                if (n < 40) {
                    if      (k < 10) v = wih[n * 10 + k];
                    else if (k < 20) v = whh[n * 10 + (k - 10)];
                }
                bf[nt][j] = (_Float16)v;
            }
            bias_n[nt] = (n < 40) ? bb[n] : 0.f;
        }

        // zero h, c (contiguous 320 floats, exact 10 strides)
        #pragma unroll
        for (int q = 0; q < 10; ++q) Hs[lane + q * 32] = 0.f;
        __syncthreads();

        for (int s = 0; s < T; ++s) {
            // A fragment, A[m][k] = k<10 ? x_t[m][k] : (k<20 ? h[m][k-10] : 0)
            // 16-bit A layout (16x32): lane<16 -> K 0..7 / 16..23 ; lane>=16 -> K 8..15 / 24..31
            v16h af;
            const float* xs = xrow + s * 10;
            if (hi == 0) {
                #pragma unroll
                for (int j = 0; j < 8; ++j) af[j] = (_Float16)xs[j];
                af[8]  = (_Float16)hrow[6];
                af[9]  = (_Float16)hrow[7];
                af[10] = (_Float16)hrow[8];
                af[11] = (_Float16)hrow[9];
                af[12] = (_Float16)0.f; af[13] = (_Float16)0.f;
                af[14] = (_Float16)0.f; af[15] = (_Float16)0.f;
            } else {
                af[0] = (_Float16)xs[8];
                af[1] = (_Float16)xs[9];
                #pragma unroll
                for (int j = 0; j < 6; ++j) af[2 + j] = (_Float16)hrow[j];
                #pragma unroll
                for (int j = 8; j < 16; ++j) af[j] = (_Float16)0.f;
            }

            // three 16-column gate tiles: G = A x B + bias (fragment-order store)
            #pragma unroll
            for (int nt = 0; nt < 3; ++nt) {
                v8f c;
                #pragma unroll
                for (int r = 0; r < 8; ++r) c[r] = bias_n[nt];
                c = __builtin_amdgcn_wmma_f32_16x16x32_f16(
                        false, af, false, bf[nt], (short)0, c, false, false);
                float4* gp = (float4*)(gfr + nt * 8);
                gp[0] = make_float4(c[0], c[1], c[2], c[3]);
                gp[1] = make_float4(c[4], c[5], c[6], c[7]);
            }
            __syncthreads();

            // cell update (addresses precomputed; 5 exact items per lane)
            const int s10 = s * 10;
            #pragma unroll
            for (int q = 0; q < 5; ++q) {
                const int p = lane + q * 32;
                float ig = G[gI[q]];
                float fg = G[gF[q]];
                float gg = G[gG[q]];
                float og = G[gO[q]];
                float cc = fsigm(fg) * Cs[p] + fsigm(ig) * ftanh(gg);
                float hh = fsigm(og) * ftanh(cc);
                Cs[p] = cc;
                Hs[p] = hh;
                xbuf[xb[q] + s10] = hh;   // next layer's input
            }
            __syncthreads();
        }

        if (l == 1) {
            #pragma unroll
            for (int q = 0; q < 5; ++q) hprev[lane + q * 32] = Hs[lane + q * 32];
            __syncthreads();
        }
    }

    #pragma unroll
    for (int q = 0; q < 5; ++q)
        out[ob[q]] = Hs[lane + q * 32] + hprev[lane + q * 32];
}

// ---------------------------------------------------------------------------
// Kernel 3: WMMA dense layer. One wave per 16x16 output tile, K chunked by 32.
// out[R][N] = act(bias[n] + sum_k A[r][k] * W[n][k]);  R multiple of 16,
// N / K tails handled with guards (zero-padded fragments).
// ---------------------------------------------------------------------------
__global__ void dense_wmma_kernel(const float* __restrict__ A,   // [R][K]
                                  const float* __restrict__ W,   // [N][K]
                                  const float* __restrict__ bias,
                                  float* __restrict__ out,       // [R][N]
                                  int K, int N, int relu)
{
    const int lane = threadIdx.x;        // 32
    const int m    = lane & 15;
    const int hi   = lane >> 4;
    const int r0   = blockIdx.x * 16;
    const int n0   = blockIdx.y * 16;
    const int col  = n0 + m;
    const bool vn  = col < N;

    v8f c;
    float bn = vn ? bias[col] : 0.f;
    #pragma unroll
    for (int r = 0; r < 8; ++r) c[r] = bn;

    const float* arow = A + (size_t)(r0 + m) * K;
    const float* wrow = W + (size_t)col * K;
    const int kb = hi * 8;
    const int kh = hi * 8;

    for (int kc = 0; kc < K; kc += 32) {
        v16h af, bf;
        if (kc + 32 <= K) {                       // full chunk, no guards
            #pragma unroll
            for (int j = 0; j < 8; ++j) {
                af[j]     = (_Float16)arow[kc + kb + j];
                af[8 + j] = (_Float16)arow[kc + 16 + kb + j];
            }
            #pragma unroll
            for (int j = 0; j < 16; ++j)
                bf[j] = (_Float16)(vn ? wrow[kc + kh + j + (j & 8)] : 0.f);
        } else {                                  // K tail, zero-pad
            #pragma unroll
            for (int j = 0; j < 8; ++j) {
                int k1 = kc + kb + j;
                int k2 = kc + 16 + kb + j;
                af[j]     = (_Float16)(k1 < K ? arow[k1] : 0.f);
                af[8 + j] = (_Float16)(k2 < K ? arow[k2] : 0.f);
            }
            #pragma unroll
            for (int j = 0; j < 16; ++j) {
                int k = kc + kh + j + (j & 8);
                bf[j] = (_Float16)((vn && k < K) ? wrow[k] : 0.f);
            }
        }
        c = __builtin_amdgcn_wmma_f32_16x16x32_f16(
                false, af, false, bf, (short)0, c, false, false);
    }

    if (vn) {
        #pragma unroll
        for (int r = 0; r < 8; ++r) {
            float v = c[r];
            if (relu) v = lrelu(v);
            out[(size_t)(r0 + hi * 8 + r) * N + col] = v;
        }
    }
}

// ---------------------------------------------------------------------------
extern "C" void kernel_launch(void* const* d_in, const int* in_sizes, int n_in,
                              void* d_out, int out_size, void* d_ws, size_t ws_size,
                              hipStream_t stream)
{
    const float* din  = (const float*)d_in[0];
    const float* W1   = (const float*)d_in[1];
    const float* b1   = (const float*)d_in[2];
    const float* W2   = (const float*)d_in[3];
    const float* b2   = (const float*)d_in[4];
    const float* W3   = (const float*)d_in[5];
    const float* b3   = (const float*)d_in[6];
    const float* fWih = (const float*)d_in[7];
    const float* fWhh = (const float*)d_in[8];
    const float* fb   = (const float*)d_in[9];
    const float* tWih = (const float*)d_in[10];
    const float* tWhh = (const float*)d_in[11];
    const float* tb   = (const float*)d_in[12];
    const float* L1W  = (const float*)d_in[13];
    const float* L1b  = (const float*)d_in[14];
    const float* L2W  = (const float*)d_in[15];
    const float* L2b  = (const float*)d_in[16];
    const float* L3W  = (const float*)d_in[17];
    const float* L3b  = (const float*)d_in[18];
    const float* L4W  = (const float*)d_in[19];
    const float* L4b  = (const float*)d_in[20];

    float* ws = (float*)d_ws;
    float* x3 = ws;                       // [4096][18][10]
    float* fv = x3 + 4096 * 180;          // [4096][10] == [64][64][10]
    float* tv = fv + 4096 * 10;           // [64][10]
    float* a1 = tv + 640;                 // [64][128]
    float* a2 = a1 + 64 * 128;            // [64][256]
    float* a3 = a2 + 64 * 256;            // [64][64]

    // 1. conv chain (one sample per block)
    conv_chain_kernel<<<4096, 128, 0, stream>>>(din, W1, b1, W2, b2, W3, b3, x3);

    // 2. flow LSTM: B=4096, T=18  (256 wave-tiles)
    size_t sh_flow = (size_t)(16 * 18 * 10 + 160 + 160 + 768 + 160) * sizeof(float);
    lstm3_kernel<<<256, 32, sh_flow, stream>>>(x3, fWih, fWhh, fb, fv, 18);

    // 3. trace LSTM: B=64 (trace idx), T=64 (flow idx)
    size_t sh_tr = (size_t)(16 * 64 * 10 + 160 + 160 + 768 + 160) * sizeof(float);
    lstm3_kernel<<<4, 32, sh_tr, stream>>>(fv, tWih, tWhh, tb, tv, 64);

    // 4. MLP head on the matrix pipe (16x16 tiles)
    dense_wmma_kernel<<<dim3(4, 8),  32, 0, stream>>>(tv, L1W, L1b, a1, 10,  128, 1);
    dense_wmma_kernel<<<dim3(4, 16), 32, 0, stream>>>(a1, L2W, L2b, a2, 128, 256, 1);
    dense_wmma_kernel<<<dim3(4, 4),  32, 0, stream>>>(a2, L3W, L3b, a3, 256, 64,  1);
    dense_wmma_kernel<<<dim3(4, 1),  32, 0, stream>>>(a3, L4W, L4b, (float*)d_out, 64, 7, 0);
}
